// SimpleLSTM_16819091931364
// MI455X (gfx1250) — matrix operand order repaired and checked by reading, not compiled, not run on hardware
//
#include <hip/hip_runtime.h>
#include <stdint.h>

// ---------------------------------------------------------------------------
// Persistent-WMMA 2-layer LSTM for MI455X (gfx1250, wave32).
// B=128, T=1024, F=256, H=512, O=128.
// - bf16 operands, f32 accumulate via v_wmma_f32_16x16x32_bf16
// - cell state c kept in VGPRs for the entire time loop
// - h exchanged via ping-pong bf16 buffers + grid-wide atomic barrier
// - all fragment traffic forced to GLOBAL address space (no flat_load)
// ---------------------------------------------------------------------------

#define B_   128
#define T_   1024
#define F_   256
#define H_   512
#define O_   128
#define NWG  32
#define TPB  256

typedef __attribute__((ext_vector_type(16))) __bf16        v16bf;
typedef __attribute__((ext_vector_type(8)))  float         v8f;
typedef __attribute__((ext_vector_type(4)))  unsigned int  u32x4;  // POD 16B

union FragBF {
  u32x4 q[2];
  v16bf v;
};

// Force GLOBAL (addrspace 1) vector loads / scalar stores so codegen emits
// global_load_b128 / global_store_* instead of flat_* (flat ties up DScnt).
__device__ __forceinline__ u32x4 ldg_q(const unsigned short* p) {
  typedef const __attribute__((address_space(1))) u32x4* gq;
  return *(gq)(unsigned long long)p;
}
__device__ __forceinline__ void stg_u16(unsigned short* p, unsigned short v) {
  typedef __attribute__((address_space(1))) unsigned short* gp;
  *(gp)(unsigned long long)p = v;
}
__device__ __forceinline__ void stg_f32(float* p, float v) {
  typedef __attribute__((address_space(1))) float* gp;
  *(gp)(unsigned long long)p = v;
}

__device__ __forceinline__ unsigned short f2bf(float f) {
  unsigned u = __float_as_uint(f);
  u += 0x7FFFu + ((u >> 16) & 1u);   // round-to-nearest-even
  return (unsigned short)(u >> 16);
}

__device__ __forceinline__ float fast_rcp(float x) {
#if __has_builtin(__builtin_amdgcn_rcpf)
  return __builtin_amdgcn_rcpf(x);
#else
  return 1.0f / x;
#endif
}

__device__ __forceinline__ float sigmoidf_(float x) {
  return fast_rcp(1.0f + __expf(-x));
}

__device__ __forceinline__ float tanhf_(float x) {
#if __has_builtin(__builtin_amdgcn_tanhf)
  return __builtin_amdgcn_tanhf(x);           // v_tanh_f32 (gfx1250 trans op)
#elif __has_builtin(__builtin_amdgcn_tanh_f32)
  return __builtin_amdgcn_tanh_f32(x);
#else
  float e = __expf(-2.0f * x);
  return (1.0f - e) * fast_rcp(1.0f + e);
#endif
}

__device__ __forceinline__ v8f wmma_bf16(const FragBF& a, const FragBF& b, v8f c) {
  return __builtin_amdgcn_wmma_f32_16x16x32_bf16(false, a.v, false, b.v,
                                                 (short)0, c, false, false);
}

// Monotonic-counter grid barrier (all NWG blocks co-resident).
__device__ __forceinline__ void grid_sync(unsigned* cnt, unsigned* gen) {
  __threadfence();
  __syncthreads();
  if (threadIdx.x == 0) {
    __hip_atomic_fetch_add(cnt, 1u, __ATOMIC_ACQ_REL, __HIP_MEMORY_SCOPE_AGENT);
    unsigned tgt = *gen + NWG;
    while (__hip_atomic_load(cnt, __ATOMIC_ACQUIRE, __HIP_MEMORY_SCOPE_AGENT) < tgt)
      __builtin_amdgcn_s_sleep(1);
  }
  __syncthreads();
  *gen += NWG;
  __threadfence();
}

// ---------------------------------------------------------------------------
// Pre-pass kernels
// ---------------------------------------------------------------------------
__global__ void cvt_bf16(const float* __restrict__ src,
                         unsigned short* __restrict__ dst, int n) {
  int i = blockIdx.x * blockDim.x + threadIdx.x;
  int stride = gridDim.x * blockDim.x;
  for (; i < n; i += stride) dst[i] = f2bf(src[i]);
}

__global__ void init_state(const float* __restrict__ h0,
                           unsigned short* __restrict__ bufA0,
                           unsigned short* __restrict__ bufB0,
                           unsigned* __restrict__ counter) {
  int i = blockIdx.x * blockDim.x + threadIdx.x;
  if (i == 0) *counter = 0u;
  int stride = gridDim.x * blockDim.x;
  for (; i < B_ * H_; i += stride) {
    int b = i / H_, h = i % H_;
    bufA0[i] = f2bf(h0[(b * 2 + 0) * H_ + h]);  // layer 0 h
    bufB0[i] = f2bf(h0[(b * 2 + 1) * H_ + h]);  // layer 1 h
  }
}

// ---------------------------------------------------------------------------
// Persistent LSTM kernel
// ---------------------------------------------------------------------------
__global__ void __launch_bounds__(TPB)
lstm_persistent(const unsigned short* __restrict__ xb,
                const unsigned short* __restrict__ Wih0b,
                const unsigned short* __restrict__ Whh0b,
                const unsigned short* __restrict__ Wih1b,
                const unsigned short* __restrict__ Whh1b,
                const unsigned short* __restrict__ Woutb,
                const float* __restrict__ bih0, const float* __restrict__ bhh0,
                const float* __restrict__ bih1, const float* __restrict__ bhh1,
                const float* __restrict__ bout,
                const float* __restrict__ c0,
                unsigned short* __restrict__ bufA0, unsigned short* __restrict__ bufA1,
                unsigned short* __restrict__ bufB0, unsigned short* __restrict__ bufB1,
                unsigned* __restrict__ counter,
                float* __restrict__ out)
{
  const int wg   = blockIdx.x;           // 0..31  -> hidden slice
  const int wave = threadIdx.x >> 5;     // 0..7   -> batch rows
  const int lane = threadIdx.x & 31;
  const int hi   = lane >> 4;            // half-wave
  const int ln   = lane & 15;
  const int j0   = wg * 16;
  const int m0   = wave * 16;

  unsigned short* bufA[2] = {bufA0, bufA1};
  unsigned short* bufB[2] = {bufB0, bufB1};

  // Per-lane biases (i,f,g,o), both layers; lane's hidden col = j0+ln.
  float b0[4], b1[4];
#pragma unroll
  for (int g = 0; g < 4; ++g) {
    b0[g] = bih0[g * H_ + j0 + ln] + bhh0[g * H_ + j0 + ln];
    b1[g] = bih1[g * H_ + j0 + ln] + bhh1[g * H_ + j0 + ln];
  }

  // Loop-invariant weight-row bases (one row of W per lane, per gate).
  const unsigned short* w0x[4], *w0h[4], *w1a[4], *w1b[4];
#pragma unroll
  for (int g = 0; g < 4; ++g) {
    const int row = g * H_ + j0 + ln;
    w0x[g] = Wih0b + (size_t)row * F_;
    w0h[g] = Whh0b + (size_t)row * H_;
    w1a[g] = Wih1b + (size_t)row * H_;
    w1b[g] = Whh1b + (size_t)row * H_;
  }

  // Loop-invariant A-row bases.
  const int arow_off = (m0 + ln) * H_;
  const unsigned short* xbase = xb + (size_t)(m0 + ln) * T_ * F_;
  const unsigned short* hAr[2] = {bufA0 + arow_off, bufA1 + arow_off};
  const unsigned short* hBr[2] = {bufB0 + arow_off, bufB1 + arow_off};

  // Cell state in VGPRs for the whole time loop (C/D tile layout:
  // VGPR r, lane -> m = m0 + r + 8*hi, n = j0 + ln).
  v8f c0r, c1r;
#pragma unroll
  for (int r = 0; r < 8; ++r) {
    int m = m0 + r + 8 * hi;
    c0r[r] = c0[(m * 2 + 0) * H_ + j0 + ln];
    c1r[r] = c0[(m * 2 + 1) * H_ + j0 + ln];
  }

  // Output-GEMM tile for waves 0,1: 64 tiles of 16x16 across 32 WGs.
  const int ytile = wg * 2 + wave;
  const int ymt   = ytile >> 3;
  const int ynt   = ytile & 7;
  const float boutr = (wave < 2) ? bout[ynt * 16 + ln] : 0.0f;
  const unsigned short* yArow0 = bufB0 + (ymt * 16 + ln) * H_;
  const unsigned short* yArow1 = bufB1 + (ymt * 16 + ln) * H_;
  const unsigned short* yWrow  = Woutb + (ynt * 16 + ln) * H_;

  const size_t hT_off = (size_t)B_ * T_ * O_;
  const size_t cT_off = hT_off + (size_t)B_ * 2 * H_;

  unsigned gen = 0;

#pragma unroll 1
  for (int t = 0; t < T_; ++t) {
    const int rd = t & 1, wr = rd ^ 1;
    unsigned short* hA_wr = bufA[wr];
    unsigned short* hB_wr = bufB[wr];

    // ---- Y(t-1) = hB(t-1) @ Wout^T + bout   (waves 0,1; overlapped) ----
    if (t > 0 && wave < 2) {
      v8f acc = {};
      const unsigned short* arow = rd ? yArow1 : yArow0;
#pragma unroll 4
      for (int kc = 0; kc < H_ / 32; ++kc) {
        const int k0 = kc * 32;
        FragBF A, Bm;
        A.q[0]  = ldg_q(arow + k0 + hi * 8);
        A.q[1]  = ldg_q(arow + k0 + 16 + hi * 8);
        Bm.q[0] = ldg_q(yWrow + k0 + hi * 8);
        Bm.q[1] = ldg_q(yWrow + k0 + 16 + hi * 8);
        acc = wmma_bf16(A, Bm, acc);
      }
#pragma unroll
      for (int r = 0; r < 8; ++r) {
        int m = ymt * 16 + r + 8 * hi;
        stg_f32(&out[((size_t)m * T_ + (t - 1)) * O_ + ynt * 16 + ln],
                acc[r] + boutr);
      }
    }

    // ---- Layer 0: gates = [x_t | hA(t-1)] @ [Wih0 | Whh0]^T, K = 768 ----
    {
      v8f acc[4] = {};
      const unsigned short* xrow = xbase + (size_t)t * F_;
      const unsigned short* hrow = hAr[rd];
#pragma unroll 2
      for (int kc = 0; kc < (F_ + H_) / 32; ++kc) {
        const int k0 = kc * 32;
        FragBF A;
        if (k0 < F_) {
          A.q[0] = ldg_q(xrow + k0 + hi * 8);
          A.q[1] = ldg_q(xrow + k0 + 16 + hi * 8);
        } else {
          const int k = k0 - F_;
          A.q[0] = ldg_q(hrow + k + hi * 8);
          A.q[1] = ldg_q(hrow + k + 16 + hi * 8);
        }
#pragma unroll
        for (int g = 0; g < 4; ++g) {
          FragBF Bm;
          if (k0 < F_) {
            Bm.q[0] = ldg_q(w0x[g] + k0 + hi * 8);
            Bm.q[1] = ldg_q(w0x[g] + k0 + 16 + hi * 8);
          } else {
            const int k = k0 - F_;
            Bm.q[0] = ldg_q(w0h[g] + k + hi * 8);
            Bm.q[1] = ldg_q(w0h[g] + k + 16 + hi * 8);
          }
          acc[g] = wmma_bf16(A, Bm, acc[g]);
        }
      }
      // LSTM cell, per lane (i,f,g,o live in the same lane slot)
#pragma unroll
      for (int r = 0; r < 8; ++r) {
        float iv = sigmoidf_(acc[0][r] + b0[0]);
        float fv = sigmoidf_(acc[1][r] + b0[1]);
        float gv = tanhf_(acc[2][r] + b0[2]);
        float ov = sigmoidf_(acc[3][r] + b0[3]);
        float cn = fv * c0r[r] + iv * gv;
        float hn = ov * tanhf_(cn);
        c0r[r] = cn;
        int m = m0 + r + 8 * hi;
        stg_u16(&hA_wr[m * H_ + j0 + ln], f2bf(hn));
        if (t == T_ - 1) {
          stg_f32(&out[hT_off + (m * 2 + 0) * H_ + j0 + ln], hn);
          stg_f32(&out[cT_off + (m * 2 + 0) * H_ + j0 + ln], cn);
        }
      }
    }
    grid_sync(counter, &gen);

    // ---- Layer 1: gates = [hA(t) | hB(t-1)] @ [Wih1 | Whh1]^T, K = 1024 ----
    {
      v8f acc[4] = {};
      const unsigned short* arow0 = hAr[wr];  // hA(t), just synced
      const unsigned short* arow1 = hBr[rd];  // hB(t-1)
#pragma unroll 2
      for (int kc = 0; kc < (2 * H_) / 32; ++kc) {
        const int k0 = kc * 32;
        FragBF A;
        if (k0 < H_) {
          A.q[0] = ldg_q(arow0 + k0 + hi * 8);
          A.q[1] = ldg_q(arow0 + k0 + 16 + hi * 8);
        } else {
          const int k = k0 - H_;
          A.q[0] = ldg_q(arow1 + k + hi * 8);
          A.q[1] = ldg_q(arow1 + k + 16 + hi * 8);
        }
#pragma unroll
        for (int g = 0; g < 4; ++g) {
          FragBF Bm;
          if (k0 < H_) {
            Bm.q[0] = ldg_q(w1a[g] + k0 + hi * 8);
            Bm.q[1] = ldg_q(w1a[g] + k0 + 16 + hi * 8);
          } else {
            const int k = k0 - H_;
            Bm.q[0] = ldg_q(w1b[g] + k + hi * 8);
            Bm.q[1] = ldg_q(w1b[g] + k + 16 + hi * 8);
          }
          acc[g] = wmma_bf16(A, Bm, acc[g]);
        }
      }
#pragma unroll
      for (int r = 0; r < 8; ++r) {
        float iv = sigmoidf_(acc[0][r] + b1[0]);
        float fv = sigmoidf_(acc[1][r] + b1[1]);
        float gv = tanhf_(acc[2][r] + b1[2]);
        float ov = sigmoidf_(acc[3][r] + b1[3]);
        float cn = fv * c1r[r] + iv * gv;
        float hn = ov * tanhf_(cn);
        c1r[r] = cn;
        int m = m0 + r + 8 * hi;
        stg_u16(&hB_wr[m * H_ + j0 + ln], f2bf(hn));
        if (t == T_ - 1) {
          stg_f32(&out[hT_off + (m * 2 + 1) * H_ + j0 + ln], hn);
          stg_f32(&out[cT_off + (m * 2 + 1) * H_ + j0 + ln], cn);
        }
      }
    }
    grid_sync(counter, &gen);
  }

  // ---- Y(T-1): hB(T-1) is in bufB[T_ & 1] ----
  if (wave < 2) {
    v8f acc = {};
    const unsigned short* arow = (T_ & 1) ? yArow1 : yArow0;
#pragma unroll 4
    for (int kc = 0; kc < H_ / 32; ++kc) {
      const int k0 = kc * 32;
      FragBF A, Bm;
      A.q[0]  = ldg_q(arow + k0 + hi * 8);
      A.q[1]  = ldg_q(arow + k0 + 16 + hi * 8);
      Bm.q[0] = ldg_q(yWrow + k0 + hi * 8);
      Bm.q[1] = ldg_q(yWrow + k0 + 16 + hi * 8);
      acc = wmma_bf16(A, Bm, acc);
    }
#pragma unroll
    for (int r = 0; r < 8; ++r) {
      int m = ymt * 16 + r + 8 * hi;
      stg_f32(&out[((size_t)m * T_ + (T_ - 1)) * O_ + ynt * 16 + ln],
              acc[r] + boutr);
    }
  }
}

// ---------------------------------------------------------------------------
// Host launcher
// ---------------------------------------------------------------------------
extern "C" void kernel_launch(void* const* d_in, const int* in_sizes, int n_in,
                              void* d_out, int out_size, void* d_ws, size_t ws_size,
                              hipStream_t stream) {
  const float* x    = (const float*)d_in[0];
  const float* h0   = (const float*)d_in[1];
  const float* c0   = (const float*)d_in[2];
  const float* Wih0 = (const float*)d_in[3];
  const float* Whh0 = (const float*)d_in[4];
  const float* bih0 = (const float*)d_in[5];
  const float* bhh0 = (const float*)d_in[6];
  const float* Wih1 = (const float*)d_in[7];
  const float* Whh1 = (const float*)d_in[8];
  const float* bih1 = (const float*)d_in[9];
  const float* bhh1 = (const float*)d_in[10];
  const float* Wout = (const float*)d_in[11];
  const float* bout = (const float*)d_in[12];
  float* out = (float*)d_out;

  char* ws = (char*)d_ws;
  size_t off = 0;
  auto carve = [&](size_t bytes) -> void* {
    void* p = ws + off;
    off += (bytes + 255) & ~(size_t)255;
    return p;
  };

  unsigned short* xb    = (unsigned short*)carve((size_t)B_ * T_ * F_ * 2);
  unsigned short* Wih0b = (unsigned short*)carve((size_t)4 * H_ * F_ * 2);
  unsigned short* Whh0b = (unsigned short*)carve((size_t)4 * H_ * H_ * 2);
  unsigned short* Wih1b = (unsigned short*)carve((size_t)4 * H_ * H_ * 2);
  unsigned short* Whh1b = (unsigned short*)carve((size_t)4 * H_ * H_ * 2);
  unsigned short* Woutb = (unsigned short*)carve((size_t)O_ * H_ * 2);
  unsigned short* bufA0 = (unsigned short*)carve((size_t)B_ * H_ * 2);
  unsigned short* bufA1 = (unsigned short*)carve((size_t)B_ * H_ * 2);
  unsigned short* bufB0 = (unsigned short*)carve((size_t)B_ * H_ * 2);
  unsigned short* bufB1 = (unsigned short*)carve((size_t)B_ * H_ * 2);
  unsigned*       cnt   = (unsigned*)carve(256);

  // f32 -> bf16 pre-conversion (x and all weight matrices)
  cvt_bf16<<<dim3(2048), dim3(TPB), 0, stream>>>(x, xb, B_ * T_ * F_);
  cvt_bf16<<<dim3(512),  dim3(TPB), 0, stream>>>(Wih0, Wih0b, 4 * H_ * F_);
  cvt_bf16<<<dim3(1024), dim3(TPB), 0, stream>>>(Whh0, Whh0b, 4 * H_ * H_);
  cvt_bf16<<<dim3(1024), dim3(TPB), 0, stream>>>(Wih1, Wih1b, 4 * H_ * H_);
  cvt_bf16<<<dim3(1024), dim3(TPB), 0, stream>>>(Whh1, Whh1b, 4 * H_ * H_);
  cvt_bf16<<<dim3(128),  dim3(TPB), 0, stream>>>(Wout, Woutb, O_ * H_);
  init_state<<<dim3(256), dim3(TPB), 0, stream>>>(h0, bufA0, bufB0, cnt);

  lstm_persistent<<<dim3(NWG), dim3(TPB), 0, stream>>>(
      xb, Wih0b, Whh0b, Wih1b, Whh1b, Woutb,
      bih0, bhh0, bih1, bhh1, bout, c0,
      bufA0, bufA1, bufB0, bufB1, cnt, out);
}
